// StackedGRU_39719857553533
// MI455X (gfx1250) — matrix operand order repaired and checked by reading, not compiled, not run on hardware
//
#include <hip/hip_runtime.h>
#include <hip/hip_bf16.h>
#include <stdint.h>

#define LNUM 4
#define BB   32
#define TT   512
#define HH   512
#define ROWS (BB * TT)   /* 16384 */
#define G3H  (3 * HH)    /* 1536  */

typedef __attribute__((ext_vector_type(16))) __bf16 v16bf;
typedef __attribute__((ext_vector_type(8)))  float  v8f;
typedef int v4i_vs __attribute__((vector_size(16)));   // builtin's pointee type

#if defined(__has_builtin)
#  if __has_builtin(__builtin_amdgcn_global_load_async_to_lds_b128)
#    define HAVE_ASYNC_B128 1
#  endif
#endif

__device__ __forceinline__ unsigned short f2bf(float x) {
  union { float f; unsigned u; } c; c.f = x;
  unsigned r = c.u + 0x7FFFu + ((c.u >> 16) & 1u);
  return (unsigned short)(r >> 16);
}

// A fragment (16x32 bf16, MxK), generic pointer (LDS or global):
// lane half h: elems 0..15 -> K = k0+16h+e
__device__ __forceinline__ v16bf load_afrag(const unsigned short* base, int stride,
                                            int m0, int k0, int lane) {
  int hh = lane >> 4, m = lane & 15;
  const unsigned short* p = base + (size_t)(m0 + m) * stride + k0 + 16 * hh;
  v16bf r;
  __builtin_memcpy(&r, p, 16);
  __builtin_memcpy(((char*)&r) + 16, p + 8, 16);
  return r;
}

// B fragment (32x16 bf16, KxN): lane l: N = l&15, Kbase = k0 + 16*(l>>4), K-contiguous
__device__ __forceinline__ v16bf load_bfrag(const unsigned short* base, int stride,
                                            int n0, int k0, int lane) {
  int hh = lane >> 4, n = lane & 15;
  const unsigned short* p = base + (size_t)(n0 + n) * stride + k0 + 16 * hh;
  v16bf r;
  __builtin_memcpy(&r, p, 32);
  return r;
}

// ---------------- LayerNorm (one wave32 per row of 512) ----------------
__global__ __launch_bounds__(256) void ln_kernel(const float* __restrict__ in,
                                                 const float* __restrict__ g,
                                                 const float* __restrict__ bta,
                                                 float* __restrict__ outf,
                                                 unsigned short* __restrict__ outb) {
  int row  = blockIdx.x * 8 + (threadIdx.x >> 5);
  int lane = threadIdx.x & 31;
  const float* p = in + (size_t)row * HH;
  float v[16];
  float s = 0.f, s2 = 0.f;
#pragma unroll
  for (int i = 0; i < 16; ++i) {
    v[i] = p[lane + 32 * i];
    s += v[i]; s2 += v[i] * v[i];
  }
#pragma unroll
  for (int off = 16; off > 0; off >>= 1) {
    s  += __shfl_xor(s, off);
    s2 += __shfl_xor(s2, off);
  }
  float mu  = s * (1.f / HH);
  float var = s2 * (1.f / HH) - mu * mu;
  float rs  = rsqrtf(var + 1e-10f);
#pragma unroll
  for (int i = 0; i < 16; ++i) {
    int c = lane + 32 * i;
    float o = (v[i] - mu) * rs * g[c] + bta[c];
    outf[(size_t)row * HH + c] = o;
    if (outb) outb[(size_t)row * HH + c] = f2bf(o);
  }
}

// ---------------- Input projection GEMM: xi = xn * W_ih^T + b_ih ----------------
// 64x64 tile per WG, full K=512 in LDS, 8 waves x 2 WMMA tiles
__global__ __launch_bounds__(256) void xi_gemm(const unsigned short* __restrict__ Ab,
                                               const float* __restrict__ W,
                                               const float* __restrict__ bias,
                                               float* __restrict__ XI) {
  extern __shared__ char smem[];
  unsigned short* As = (unsigned short*)smem;  // 64 x (512+8)
  unsigned short* Ws = As + 64 * 520;          // 64 x (512+8)
  int tid = threadIdx.x, lane = tid & 31, w = tid >> 5;
  int n0 = blockIdx.x * 64;
  int m0 = blockIdx.y * 64;

#ifdef HAVE_ASYNC_B128
  // CDNA5 async global->LDS copy (bypasses VGPRs), tracked by ASYNCcnt
  for (int c = tid; c < 64 * 64; c += 256) {
    int r = c >> 6, cc = (c & 63) * 8;
    __builtin_amdgcn_global_load_async_to_lds_b128(
        (v4i_vs*)&Ab[(size_t)(m0 + r) * HH + cc],
        (v4i_vs*)&As[r * 520 + cc],
        0, 0);
  }
#else
  for (int c = tid; c < 64 * 64; c += 256) {   // A: bf16 copy, 16B chunks
    int r = c >> 6, cc = (c & 63) * 8;
    *(uint4*)&As[r * 520 + cc] = *(const uint4*)&Ab[(size_t)(m0 + r) * HH + cc];
  }
#endif
  for (int c = tid; c < 64 * 128; c += 256) {  // W: fp32 -> bf16
    int r = c >> 7, cc = (c & 127) * 4;
    float4 wv = *(const float4*)&W[(size_t)(n0 + r) * HH + cc];
    unsigned short* d = &Ws[r * 520 + cc];
    d[0] = f2bf(wv.x); d[1] = f2bf(wv.y); d[2] = f2bf(wv.z); d[3] = f2bf(wv.w);
  }
#ifdef HAVE_ASYNC_B128
#  if __has_builtin(__builtin_amdgcn_s_wait_asynccnt)
  __builtin_amdgcn_s_wait_asynccnt(0);
#  else
  asm volatile("s_wait_asynccnt 0x0" ::: "memory");
#  endif
#endif
  __syncthreads();

#pragma unroll
  for (int tsel = 0; tsel < 2; ++tsel) {
    int tt = w + tsel * 8;
    int mt = tt >> 2, nt = tt & 3;
    v8f acc = {0.f, 0.f, 0.f, 0.f, 0.f, 0.f, 0.f, 0.f};
#pragma unroll
    for (int ks = 0; ks < 16; ++ks) {
      v16bf af = load_afrag(As, 520, 16 * mt, 32 * ks, lane);
      v16bf bf = load_bfrag(Ws, 520, 16 * nt, 32 * ks, lane);
      acc = __builtin_amdgcn_wmma_f32_16x16x32_bf16(false, af, false, bf,
                                                    (short)0, acc, false, false);
    }
    int n = lane & 15, hf = lane >> 4;
    int gn = n0 + 16 * nt + n;
    float bi = bias[gn];
#pragma unroll
    for (int vv = 0; vv < 8; ++vv) {
      int m = m0 + 16 * mt + vv + 8 * hf;
      XI[(size_t)m * G3H + gn] = acc[vv] + bi;
    }
  }
}

// ---------------- grid-wide sync (persistent cooperative scan) ----------------
__device__ __forceinline__ void grid_sync(volatile unsigned* cnt, unsigned target) {
  __threadfence();
  __syncthreads();
  if (threadIdx.x == 0) {
    atomicAdd((unsigned*)cnt, 1u);
    while (*cnt < target) __builtin_amdgcn_s_sleep(2);
  }
  __syncthreads();
  __threadfence();
}

__global__ void zero_kernel(unsigned* p) { if (threadIdx.x == 0) *p = 0u; }

// ---------------- Recurrent GRU scan: 32 blocks, each owns 16 hidden cols ----------------
__global__ __launch_bounds__(256) void gru_scan(const float* __restrict__ XI,
                                                const float* __restrict__ XN,
                                                const float* __restrict__ Whh,
                                                const float* __restrict__ bhh,
                                                const float* __restrict__ h0l,
                                                unsigned short* __restrict__ HB,  // 2 x [32x512] bf16 ping-pong
                                                float* __restrict__ OUTC,         // [B,T,H] residual out
                                                float* __restrict__ HOUT,         // [B,H] last hidden
                                                unsigned* __restrict__ syncp) {
  extern __shared__ char smem[];
  unsigned short* lds_w = (unsigned short*)smem;  // 48 x 520 bf16 (resident W_hh slice)
  float* lds_gh = (float*)(lds_w + 48 * 520);     // 32 x 52
  float* lds_hf = lds_gh + 32 * 52;               // 32 x 16 fp32 persistent h (own cols)
  float* lds_bh = lds_hf + 32 * 16;               // 48
  int tid = threadIdx.x, lane = tid & 31, w = tid >> 5;
  int blk = blockIdx.x;  // owns hidden columns [16*blk, 16*blk+16)

  // init: stage W_hh rows for r/z/n gates of owned columns, fp32 -> bf16
  for (int c = tid; c < 48 * 128; c += 256) {
    int r = c >> 7, cc = (c & 127) * 4;
    int gate = r >> 4, j = r & 15;
    float4 wv = *(const float4*)&Whh[(size_t)(512 * gate + 16 * blk + j) * HH + cc];
    unsigned short* d = &lds_w[r * 520 + cc];
    d[0] = f2bf(wv.x); d[1] = f2bf(wv.y); d[2] = f2bf(wv.z); d[3] = f2bf(wv.w);
  }
  if (tid < 48) {
    int gate = tid >> 4, j = tid & 15;
    lds_bh[tid] = bhh[512 * gate + 16 * blk + j];
  }
  for (int p = tid; p < 512; p += 256) {  // h0 -> persistent fp32 + bf16 ping-pong buf 0
    int b = p >> 4, j = p & 15;
    float h = h0l[(size_t)b * HH + 16 * blk + j];
    lds_hf[b * 16 + j] = h;
    HB[(size_t)b * HH + 16 * blk + j] = f2bf(h);
  }
  grid_sync(syncp, 32u);

  // each thread handles two (b, j) gate pairs: p0 = tid, p1 = tid + 256
  int b0 = tid >> 4,         j0 = tid & 15,         col0 = 16 * blk + j0;
  int b1 = (tid + 256) >> 4, j1 = (tid + 256) & 15, col1 = 16 * blk + j1;

  for (int t = 0; t < TT; ++t) {
    const unsigned short* hb = HB + (size_t)(t & 1) * (BB * HH);

    // issue gate-input loads early so their latency hides under the WMMA work
    size_t xib0 = ((size_t)b0 * TT + t) * G3H;
    size_t xib1 = ((size_t)b1 * TT + t) * G3H;
    float xr0 = XI[xib0 + col0], xz0 = XI[xib0 + 512 + col0], xg0 = XI[xib0 + 1024 + col0];
    float xr1 = XI[xib1 + col1], xz1 = XI[xib1 + 512 + col1], xg1 = XI[xib1 + 1024 + col1];
    if (t + 1 < TT) {  // prefetch next step's gate inputs (global_prefetch_b8)
      __builtin_prefetch(&XI[xib0 + G3H + col0], 0, 0);
      __builtin_prefetch(&XI[xib0 + G3H + 512 + col0], 0, 0);
      __builtin_prefetch(&XI[xib0 + G3H + 1024 + col0], 0, 0);
      __builtin_prefetch(&XI[xib1 + G3H + col1], 0, 0);
      __builtin_prefetch(&XI[xib1 + G3H + 512 + col1], 0, 0);
      __builtin_prefetch(&XI[xib1 + G3H + 1024 + col1], 0, 0);
    }

    if (w < 6) {  // 2 M-tiles x 3 gate-tiles, K = 512; A direct from global ping-pong
      int mt = w & 1, gate = w >> 1;
      v8f acc = {0.f, 0.f, 0.f, 0.f, 0.f, 0.f, 0.f, 0.f};
#pragma unroll
      for (int ks = 0; ks < 16; ++ks) {
        v16bf af = load_afrag(hb, HH, 16 * mt, 32 * ks, lane);        // global_load_b128 x2
        v16bf bf = load_bfrag(lds_w, 520, 16 * gate, 32 * ks, lane);  // ds_load
        acc = __builtin_amdgcn_wmma_f32_16x16x32_bf16(false, af, false, bf,
                                                      (short)0, acc, false, false);
      }
      int n = lane & 15, hf = lane >> 4;
#pragma unroll
      for (int vv = 0; vv < 8; ++vv) {
        int m = 16 * mt + vv + 8 * hf;
        lds_gh[m * 52 + 16 * gate + n] = acc[vv];
      }
    }
    __syncthreads();

    unsigned short* hbn = HB + (size_t)((t + 1) & 1) * (BB * HH);
    {
      float ghr = lds_gh[b0 * 52 + j0]      + lds_bh[j0];
      float ghz = lds_gh[b0 * 52 + 16 + j0] + lds_bh[16 + j0];
      float ghn = lds_gh[b0 * 52 + 32 + j0] + lds_bh[32 + j0];
      float r  = 1.f / (1.f + __expf(-(xr0 + ghr)));
      float z  = 1.f / (1.f + __expf(-(xz0 + ghz)));
      float ng = tanhf(xg0 + r * ghn);
      float hp = lds_hf[b0 * 16 + j0];
      float hn = (1.f - z) * ng + z * hp;
      lds_hf[b0 * 16 + j0] = hn;
      hbn[(size_t)b0 * HH + col0] = f2bf(hn);
      size_t oi = ((size_t)b0 * TT + t) * HH + col0;
      OUTC[oi] = hn + XN[oi];
      if (t == TT - 1) HOUT[(size_t)b0 * HH + col0] = hn;
    }
    {
      float ghr = lds_gh[b1 * 52 + j1]      + lds_bh[j1];
      float ghz = lds_gh[b1 * 52 + 16 + j1] + lds_bh[16 + j1];
      float ghn = lds_gh[b1 * 52 + 32 + j1] + lds_bh[32 + j1];
      float r  = 1.f / (1.f + __expf(-(xr1 + ghr)));
      float z  = 1.f / (1.f + __expf(-(xz1 + ghz)));
      float ng = tanhf(xg1 + r * ghn);
      float hp = lds_hf[b1 * 16 + j1];
      float hn = (1.f - z) * ng + z * hp;
      lds_hf[b1 * 16 + j1] = hn;
      hbn[(size_t)b1 * HH + col1] = f2bf(hn);
      size_t oi = ((size_t)b1 * TT + t) * HH + col1;
      OUTC[oi] = hn + XN[oi];
      if (t == TT - 1) HOUT[(size_t)b1 * HH + col1] = hn;
    }
    grid_sync(syncp, 32u * (unsigned)(t + 2));
  }
}

extern "C" void kernel_launch(void* const* d_in, const int* in_sizes, int n_in,
                              void* d_out, int out_size, void* d_ws, size_t ws_size,
                              hipStream_t stream) {
  (void)in_sizes; (void)n_in; (void)out_size; (void)ws_size;
  const float* x     = (const float*)d_in[0];
  const float* h0    = (const float*)d_in[1];
  const float* Wih   = (const float*)d_in[2];
  const float* Whh   = (const float*)d_in[3];
  const float* bih   = (const float*)d_in[4];
  const float* bhh   = (const float*)d_in[5];
  const float* lng   = (const float*)d_in[6];
  const float* lnb   = (const float*)d_in[7];
  const float* lastg = (const float*)d_in[8];
  const float* lastb = (const float*)d_in[9];

  float* out  = (float*)d_out;                    // [B,T,H] then [L,B,H]
  float* outh = out + (size_t)ROWS * HH;

  // workspace layout
  float* A  = (float*)d_ws;                       // cur buffer   [ROWS, H]
  float* XN = A + (size_t)ROWS * HH;              // layernorm'd  [ROWS, H]
  float* XI = XN + (size_t)ROWS * HH;             // projections  [ROWS, 3H]
  unsigned short* XNB = (unsigned short*)(XI + (size_t)ROWS * G3H);  // bf16 xn
  unsigned short* HB  = XNB + (size_t)ROWS * HH;  // 2 x [B,H] bf16 ping-pong
  unsigned* SYNC = (unsigned*)(HB + 2 * BB * HH);

  size_t smem_xi  = (size_t)(2 * 64 * 520) * sizeof(unsigned short);   // 133120 B
  size_t smem_rec = (size_t)(48 * 520) * 2 +
                    (size_t)(32 * 52 + 32 * 16 + 48) * 4;              // 58816 B

  for (int l = 0; l < LNUM; ++l) {
    const float* curIn = (l == 0) ? x : A;
    ln_kernel<<<dim3(ROWS / 8), dim3(256), 0, stream>>>(
        curIn, lng + (size_t)l * HH, lnb + (size_t)l * HH, XN, XNB);
    xi_gemm<<<dim3(G3H / 64, ROWS / 64), dim3(256), smem_xi, stream>>>(
        XNB, Wih + (size_t)l * G3H * HH, bih + (size_t)l * G3H, XI);
    zero_kernel<<<dim3(1), dim3(32), 0, stream>>>(SYNC);
    gru_scan<<<dim3(32), dim3(256), smem_rec, stream>>>(
        XI, XN, Whh + (size_t)l * G3H * HH, bhh + (size_t)l * G3H,
        h0 + (size_t)l * BB * HH, HB, A, outh + (size_t)l * BB * HH, SYNC);
  }
  ln_kernel<<<dim3(ROWS / 8), dim3(256), 0, stream>>>(
      A, lastg, lastb, out, (unsigned short*)nullptr);
}
